// SwinDeformableAttention_52286931862249
// MI455X (gfx1250) — compile-verified, matrix operand order checked by reading, not run on hardware
//
#include <hip/hip_runtime.h>
#include <hip/hip_bf16.h>
#include <cstdint>

// ---------------------------------------------------------------------------
// SwinDeformableAttention on gfx1250 (MI455X): flash-attention streaming with
// v_wmma_f32_16x16x32_f16 for all matmuls, and async global->LDS double
// buffering (global_load_async_to_lds_b128 / s_wait_asynccnt via inline asm)
// of the K/V tiles in the attention hot loop. wave32; one wave per 16-row tile.
// ---------------------------------------------------------------------------

typedef __attribute__((ext_vector_type(16))) _Float16 v16h;
typedef __attribute__((ext_vector_type(8)))  float    v8f;

#define BB   4
#define CC   64
#define HH   40
#define WW   40
#define NN   1600           // H*W
#define KP   4
#define KN   6400           // KP*NN
#define HWX  1600           // H*W (plane size)
#define CH   (KN / 32)      // 200 key chunks

#if defined(__gfx1250__)
#define SDA_ASYNC 1
#else
#define SDA_ASYNC 0
#endif

// 16-bit A-matrix 16x32 lane/element -> K mapping (CDNA5 ISA 7.12.2).
// e=0..7 -> K=0..7 (+8*hi), e=8..15 -> K=16..23 (+8*hi): two contiguous
// 8-element (16-byte) runs -> vectorizes to (global|ds)_load_b128.
__device__ __forceinline__ int a_k_of(int lane, int e) {
    int v = e >> 1, h = e & 1;
    int kk = (v < 4) ? (2 * v + h) : (16 + 2 * (v - 4) + h);
    return kk + ((lane & 16) ? 8 : 0);
}

__device__ __forceinline__ v8f wmma_f16(v16h a, v16h b, v8f c) {
    return __builtin_amdgcn_wmma_f32_16x16x32_f16(false, a, false, b,
                                                  (short)0, c, false, false);
}

// 16x32 f16 operand from a row-major f16 matrix (global or LDS).
__device__ __forceinline__ v16h load_a_f16(const _Float16* M, long rowBase,
                                           int ld, int k0, int lane) {
    v16h a;
    const _Float16* p = M + (rowBase + (lane & 15)) * (long)ld + k0;
#pragma unroll
    for (int e = 0; e < 16; ++e) a[e] = p[a_k_of(lane, e)];
    return a;
}

// B (32x16 f16) for X @ W^T: B[k,n] = W[nBase+n][k0+k], W row-major f32 [n][k]
__device__ __forceinline__ v16h load_b_wt_f32(const float* Wt, int nBase,
                                              int ld, int k0, int lane) {
    v16h b;
    const float* p = Wt + (long)(nBase + (lane & 15)) * ld + k0;
#pragma unroll
    for (int e = 0; e < 16; ++e) b[e] = (_Float16)p[a_k_of(lane, e)];
    return b;
}

// async 16-byte global -> LDS copy (per-lane), ASYNCcnt-tracked.
// LDS operand VGPR holds the wave-relative LDS byte offset: for a generic
// pointer derived from a __shared__ object that is its low 32 bits.
__device__ __forceinline__ void async_copy16(const void* g, void* l) {
#if SDA_ASYNC
    const uint32_t loff = (uint32_t)(uintptr_t)l;
    const uint64_t ga   = (uint64_t)(uintptr_t)g;
    asm volatile("global_load_async_to_lds_b128 %0, %1, off"
                 :: "v"(loff), "v"(ga) : "memory");
#else
    *(int4*)l = *(const int4*)g;
#endif
}
__device__ __forceinline__ void async_wait_prev16() {   // <=16 outstanding
#if SDA_ASYNC
    asm volatile("s_wait_asynccnt 16" ::: "memory");
#endif
}
__device__ __forceinline__ void async_wait_all() {
#if SDA_ASYNC
    asm volatile("s_wait_asynccnt 0" ::: "memory");
#endif
}

// reductions across the 16 lanes sharing one row-group
__device__ __forceinline__ float redmax16(float v) {
    v = fmaxf(v, __shfl_xor(v, 1, 32));
    v = fmaxf(v, __shfl_xor(v, 2, 32));
    v = fmaxf(v, __shfl_xor(v, 4, 32));
    v = fmaxf(v, __shfl_xor(v, 8, 32));
    return v;
}
__device__ __forceinline__ float redsum16(float v) {
    v += __shfl_xor(v, 1, 32);
    v += __shfl_xor(v, 2, 32);
    v += __shfl_xor(v, 4, 32);
    v += __shfl_xor(v, 8, 32);
    return v;
}

// ---------------------------------------------------------------------------
// 1) Q = tokens @ Wq^T + bq; emit Qh = f16(Q*scale) and sig = row mean.
// ---------------------------------------------------------------------------
__global__ __launch_bounds__(32)
void sda_qproj(const float* __restrict__ x, const float* __restrict__ Wq,
               const float* __restrict__ bq, _Float16* __restrict__ Qh,
               float* __restrict__ sig) {
    const int lane = threadIdx.x;
    const long row0 = (long)blockIdx.x * 16;
    const int b = (int)(row0 / NN);
    const int hi = lane >> 4, nlo = lane & 15;

    v16h a0, a1;
    {
        const float* xb = x + (long)b * CC * HWX + (int)(row0 % NN) + nlo;
#pragma unroll
        for (int e = 0; e < 16; ++e) {
            a0[e] = (_Float16)xb[(long)(a_k_of(lane, e)) * HWX];
            a1[e] = (_Float16)xb[(long)(32 + a_k_of(lane, e)) * HWX];
        }
    }

    float rs[8];
#pragma unroll
    for (int r = 0; r < 8; ++r) rs[r] = 0.f;

#pragma unroll
    for (int nc = 0; nc < 4; ++nc) {
        v8f acc = {};
        acc = wmma_f16(a0, load_b_wt_f32(Wq, nc * 16, CC, 0,  lane), acc);
        acc = wmma_f16(a1, load_b_wt_f32(Wq, nc * 16, CC, 32, lane), acc);
        const float bj = bq[nc * 16 + nlo];
#pragma unroll
        for (int r = 0; r < 8; ++r) {
            const float q = acc[r] + bj;
            rs[r] += q;
            Qh[(row0 + r + 8 * hi) * CC + nc * 16 + nlo] = (_Float16)(q * 0.125f);
        }
    }
#pragma unroll
    for (int r = 0; r < 8; ++r) {
        const float s = redsum16(rs[r]) * (1.0f / 64.0f);
        if (nlo == 0) sig[row0 + r + 8 * hi] = s;
    }
}

// ---------------------------------------------------------------------------
// 2) offset = conv1d(sig, w, b) (3-tap, pad 1) -> ws copy + d_out tail
// ---------------------------------------------------------------------------
__global__ void sda_conv1d(const float* __restrict__ sig, const float* __restrict__ cw,
                           const float* __restrict__ cb, float* __restrict__ offs,
                           float* __restrict__ out_off) {
    const int idx = blockIdx.x * blockDim.x + threadIdx.x;
    if (idx >= BB * 8 * NN) return;
    const int n  = idx % NN;
    const int oc = (idx / NN) % 8;
    const int b  = idx / (8 * NN);
    const float* s = sig + (long)b * NN;
    const float left  = (n > 0)      ? s[n - 1] : 0.f;
    const float mid   = s[n];
    const float right = (n < NN - 1) ? s[n + 1] : 0.f;
    const float v = cb[oc] + cw[oc * 3 + 0] * left + cw[oc * 3 + 1] * mid
                           + cw[oc * 3 + 2] * right;
    offs[idx]    = v;
    out_off[idx] = v;
}

// ---------------------------------------------------------------------------
// 3) xs = grid_sample(x, gx, gy) bilinear, align_corners, zero pad -> f16
// ---------------------------------------------------------------------------
__global__ void sda_sample(const float* __restrict__ x, const float* __restrict__ offs,
                           _Float16* __restrict__ xsh) {
    const long idx = (long)blockIdx.x * blockDim.x + threadIdx.x;
    if (idx >= (long)BB * KN * CC) return;
    const int c = (int)(idx % CC);
    const int p = (int)((idx / CC) % KN);
    const int b = (int)(idx / ((long)CC * KN));
    const int kpt = p / NN, n = p % NN;
    const float gy = offs[((long)b * 8 + 2 * kpt + 0) * NN + n];
    const float gx = offs[((long)b * 8 + 2 * kpt + 1) * NN + n];
    const float ix = (gx + 1.0f) * 0.5f * (WW - 1);
    const float iy = (gy + 1.0f) * 0.5f * (HH - 1);
    const float ix0f = floorf(ix), iy0f = floorf(iy);
    const int ix0 = (int)ix0f, iy0 = (int)iy0f;
    const float wx1 = ix - ix0f, wx0 = 1.f - wx1;
    const float wy1 = iy - iy0f, wy0 = 1.f - wy1;
    const float* xc = x + ((long)b * CC + c) * HWX;
    float v = 0.f;
    if (ix0 >= 0 && ix0 < WW && iy0 >= 0 && iy0 < HH)         v += xc[iy0 * WW + ix0] * wy0 * wx0;
    if (ix0 + 1 >= 0 && ix0 + 1 < WW && iy0 >= 0 && iy0 < HH) v += xc[iy0 * WW + ix0 + 1] * wy0 * wx1;
    if (ix0 >= 0 && ix0 < WW && iy0 + 1 >= 0 && iy0 + 1 < HH) v += xc[(iy0 + 1) * WW + ix0] * wy1 * wx0;
    if (ix0 + 1 >= 0 && ix0 + 1 < WW && iy0 + 1 >= 0 && iy0 + 1 < HH)
        v += xc[(iy0 + 1) * WW + ix0 + 1] * wy1 * wx1;
    xsh[(((long)b * KN) + p) * CC + c] = (_Float16)v;
}

// ---------------------------------------------------------------------------
// 4) K = xs@Wk^T+bk (row-major), V = xs@Wv^T+bv stored TRANSPOSED Vt[b][c][key]
// ---------------------------------------------------------------------------
__global__ __launch_bounds__(32)
void sda_kv(const _Float16* __restrict__ xsh, const float* __restrict__ Wk,
            const float* __restrict__ bk, const float* __restrict__ Wv,
            const float* __restrict__ bv, _Float16* __restrict__ Kmh,
            _Float16* __restrict__ Vt) {
    const int lane = threadIdx.x;
    const long row0 = (long)blockIdx.x * 16;
    const int b = (int)(row0 / KN);
    const int key0 = (int)(row0 % KN);
    const int hi = lane >> 4, nlo = lane & 15;
    const v16h a0 = load_a_f16(xsh, row0, CC, 0,  lane);
    const v16h a1 = load_a_f16(xsh, row0, CC, 32, lane);
#pragma unroll
    for (int nc = 0; nc < 4; ++nc) {
        v8f k = {}, v = {};
        k = wmma_f16(a0, load_b_wt_f32(Wk, nc * 16, CC, 0,  lane), k);
        k = wmma_f16(a1, load_b_wt_f32(Wk, nc * 16, CC, 32, lane), k);
        v = wmma_f16(a0, load_b_wt_f32(Wv, nc * 16, CC, 0,  lane), v);
        v = wmma_f16(a1, load_b_wt_f32(Wv, nc * 16, CC, 32, lane), v);
        const float bkj = bk[nc * 16 + nlo], bvj = bv[nc * 16 + nlo];
        const int c = nc * 16 + nlo;
#pragma unroll
        for (int r = 0; r < 8; ++r) {
            const int key = key0 + r + 8 * hi;
            Kmh[((long)b * KN + key) * CC + c] = (_Float16)(k[r] + bkj);
            Vt [((long)b * CC + c) * KN + key] = (_Float16)(v[r] + bvj);
        }
    }
}

// ---------------------------------------------------------------------------
// 5) Flash attention over KN keys + fused output projection, with async
//    double-buffered K/V tiles in LDS. One wave per 16-query tile.
// ---------------------------------------------------------------------------

// stage one 32-key chunk: K tile (32x64 f16, contiguous 4KB) and
// V tile (64 c-rows x 32 keys f16) -> 16 async b128 issues per wave.
__device__ __forceinline__ void sda_fetch_tile(const _Float16* __restrict__ Kmh,
                                               const _Float16* __restrict__ Vt,
                                               long krow0, long vcol0, int key0,
                                               _Float16* kb, _Float16* vb, int lane) {
    const _Float16* kg = Kmh + (krow0 + key0) * CC;
#pragma unroll
    for (int i = 0; i < 8; ++i) {
        const int seg = i * 32 + lane;                   // 0..255 16B segments
        async_copy16(kg + seg * 8, kb + seg * 8);
    }
#pragma unroll
    for (int i = 0; i < 8; ++i) {
        const int seg = i * 32 + lane;                   // 0..255
        const int c = seg >> 2, s = (seg & 3) * 8;       // 64 rows x 4 segs
        async_copy16(Vt + (vcol0 + c) * KN + key0 + s, vb + c * 32 + s);
    }
}

__global__ __launch_bounds__(32)
void sda_attn(const _Float16* __restrict__ Qh, const _Float16* __restrict__ Kmh,
              const _Float16* __restrict__ Vt, const float* __restrict__ Wo,
              const float* __restrict__ bo, float* __restrict__ out) {
    __shared__ __align__(16) _Float16 kbuf[2][32 * 64];   // 2 x 4KB
    __shared__ __align__(16) _Float16 vbuf[2][64 * 32];   // 2 x 4KB
    __shared__ __align__(16) _Float16 pstage[16 * 32];    // 1KB
    const int lane = threadIdx.x;
    const int tile = blockIdx.x;
    const int b = tile / (NN / 16);
    const int n0 = (tile % (NN / 16)) * 16;
    const long qrow0 = (long)b * NN + n0;
    const long krow0 = (long)b * KN;
    const long vcol0 = (long)b * CC;
    const int hi = lane >> 4, nlo = lane & 15;

    const v16h aq0 = load_a_f16(Qh, qrow0, CC, 0,  lane);
    const v16h aq1 = load_a_f16(Qh, qrow0, CC, 32, lane);

    v8f O0 = {}, O1 = {}, O2 = {}, O3 = {};
    float mrow[8], lrow[8];
#pragma unroll
    for (int r = 0; r < 8; ++r) { mrow[r] = -1e30f; lrow[r] = 0.f; }

    sda_fetch_tile(Kmh, Vt, krow0, vcol0, 0, kbuf[0], vbuf[0], lane);

    for (int kc = 0; kc < CH; ++kc) {
        const int buf = kc & 1;
        if (kc + 1 < CH) {
            sda_fetch_tile(Kmh, Vt, krow0, vcol0, (kc + 1) * 32,
                           kbuf[buf ^ 1], vbuf[buf ^ 1], lane);
            async_wait_prev16();     // completion is in-order: tile kc resident
        } else {
            async_wait_all();
        }
        __syncthreads();             // compiler ordering (single wave: trivial)
        const _Float16* kb = kbuf[buf];
        const _Float16* vb = vbuf[buf];

        // S = (Q*scale) @ K^T : two 16x16 f32 tiles covering 32 keys
        v8f S0 = {}, S1 = {};
        S0 = wmma_f16(aq0, load_a_f16(kb, 0,  CC, 0,  lane), S0);
        S0 = wmma_f16(aq1, load_a_f16(kb, 0,  CC, 32, lane), S0);
        S1 = wmma_f16(aq0, load_a_f16(kb, 16, CC, 0,  lane), S1);
        S1 = wmma_f16(aq1, load_a_f16(kb, 16, CC, 32, lane), S1);

        float alpha[8];
#pragma unroll
        for (int r = 0; r < 8; ++r) {
            const float mx = redmax16(fmaxf(S0[r], S1[r]));
            const float mnew = fmaxf(mrow[r], mx);
            alpha[r] = __expf(mrow[r] - mnew);
            mrow[r] = mnew;
            const float p0 = __expf(S0[r] - mnew);
            const float p1 = __expf(S1[r] - mnew);
            S0[r] = p0; S1[r] = p1;
            lrow[r] = lrow[r] * alpha[r] + redsum16(p0 + p1);
        }
#pragma unroll
        for (int r = 0; r < 8; ++r) {
            O0[r] *= alpha[r]; O1[r] *= alpha[r];
            O2[r] *= alpha[r]; O3[r] *= alpha[r];
        }
        // re-stage P (D-layout) -> A-layout through LDS
#pragma unroll
        for (int r = 0; r < 8; ++r) {
            const int m = r + 8 * hi;
            pstage[m * 32 + nlo]      = (_Float16)S0[r];
            pstage[m * 32 + 16 + nlo] = (_Float16)S1[r];
        }
        __syncthreads();
        v16h ap;
        {
            const _Float16* p = pstage + nlo * 32;
#pragma unroll
            for (int e = 0; e < 16; ++e) ap[e] = p[a_k_of(lane, e)];
        }
        __syncthreads();
        // O += P @ V : B[k,n] = vb[n][k] (c-major rows, contiguous along k)
        O0 = wmma_f16(ap, load_a_f16(vb, 0,  32, 0, lane), O0);
        O1 = wmma_f16(ap, load_a_f16(vb, 16, 32, 0, lane), O1);
        O2 = wmma_f16(ap, load_a_f16(vb, 32, 32, 0, lane), O2);
        O3 = wmma_f16(ap, load_a_f16(vb, 48, 32, 0, lane), O3);
    }

    // normalize + stage O as f16 16x64 (reuse vbuf[0]) for output projection
    _Float16* ost = vbuf[0];
#pragma unroll
    for (int r = 0; r < 8; ++r) {
        const int m = r + 8 * hi;
        const float inv = 1.0f / lrow[r];
        ost[m * 64 + 0  + nlo] = (_Float16)(O0[r] * inv);
        ost[m * 64 + 16 + nlo] = (_Float16)(O1[r] * inv);
        ost[m * 64 + 32 + nlo] = (_Float16)(O2[r] * inv);
        ost[m * 64 + 48 + nlo] = (_Float16)(O3[r] * inv);
    }
    __syncthreads();
    v16h ao0, ao1;
    {
        const _Float16* p = ost + nlo * 64;
#pragma unroll
        for (int e = 0; e < 16; ++e) {
            ao0[e] = p[a_k_of(lane, e)];
            ao1[e] = p[32 + a_k_of(lane, e)];
        }
    }
    // out = O @ Wo^T + bo
#pragma unroll
    for (int nc = 0; nc < 4; ++nc) {
        v8f f = {};
        f = wmma_f16(ao0, load_b_wt_f32(Wo, nc * 16, CC, 0,  lane), f);
        f = wmma_f16(ao1, load_b_wt_f32(Wo, nc * 16, CC, 32, lane), f);
        const float bj = bo[nc * 16 + nlo];
#pragma unroll
        for (int r = 0; r < 8; ++r)
            out[(qrow0 + r + 8 * hi) * CC + nc * 16 + nlo] = f[r] + bj;
    }
}

// ---------------------------------------------------------------------------
extern "C" void kernel_launch(void* const* d_in, const int* in_sizes, int n_in,
                              void* d_out, int out_size, void* d_ws, size_t ws_size,
                              hipStream_t stream) {
    (void)in_sizes; (void)n_in; (void)out_size; (void)ws_size;
    const float* x      = (const float*)d_in[0];
    const float* Wq     = (const float*)d_in[1];
    const float* bq     = (const float*)d_in[2];
    const float* Wk     = (const float*)d_in[3];
    const float* bk     = (const float*)d_in[4];
    const float* Wv     = (const float*)d_in[5];
    const float* bv     = (const float*)d_in[6];
    const float* Wo     = (const float*)d_in[7];
    const float* bo     = (const float*)d_in[8];
    const float* conv_w = (const float*)d_in[9];
    const float* conv_b = (const float*)d_in[10];

    float* out_main = (float*)d_out;                       // [B,N,C]
    float* out_off  = out_main + (long)BB * NN * CC;       // [B,8,N]

    char* w = (char*)d_ws;
    float* sig = (float*)w;            w += sizeof(float) * BB * NN;
    float* offs = (float*)w;           w += sizeof(float) * BB * 8 * NN;
    _Float16* Qh  = (_Float16*)w;      w += sizeof(_Float16) * (long)BB * NN * CC;
    _Float16* xsh = (_Float16*)w;      w += sizeof(_Float16) * (long)BB * KN * CC;
    _Float16* Kmh = (_Float16*)w;      w += sizeof(_Float16) * (long)BB * KN * CC;
    _Float16* Vt  = (_Float16*)w;      // [B][C][KN] halves (total ~10.9 MB)

    sda_qproj <<<(BB * NN) / 16, 32, 0, stream>>>(x, Wq, bq, Qh, sig);
    sda_conv1d<<<(BB * 8 * NN + 255) / 256, 256, 0, stream>>>(sig, conv_w, conv_b,
                                                              offs, out_off);
    sda_sample<<<(int)(((long)BB * KN * CC + 255) / 256), 256, 0, stream>>>(x, offs, xsh);
    sda_kv    <<<(BB * KN) / 16, 32, 0, stream>>>(xsh, Wk, bk, Wv, bv, Kmh, Vt);
    sda_attn  <<<(BB * NN) / 16, 32, 0, stream>>>(Qh, Kmh, Vt, Wo, bo, out_main);
}
